// AttnGate_84902913507570
// MI455X (gfx1250) — matrix operand order repaired and checked by reading, not compiled, not run on hardware
//
#include <hip/hip_runtime.h>
#include <hip/hip_bf16.h>

// Problem constants (from reference)
#define BATCH 2
#define SEQ   8192
#define HQ    32
#define HK    8
#define DH    128
#define GH    128
#define NB    128
#define BLKK  64
#define GRP   4

typedef __attribute__((ext_vector_type(16))) _Float16 v16h;
typedef __attribute__((ext_vector_type(8)))  float    v8f;

union Frag { v16h v; _Float16 h[16]; };

#define A_STR 520   // 512 K + 8 pad (halves) : avoids LDS bank conflicts
#define Q_STR 136   // 128 K + 8 pad (16B-aligned rows: 272B)

// Async global->LDS copy of 16 bytes (CDNA5 GLOBAL_LOAD_ASYNC_TO_LDS_B128).
// LDS address = low 32 bits of the flat shared-aperture pointer (ISA 10.2).
__device__ __forceinline__ void async_copy_b128(void* lds_dst, const void* gsrc) {
  unsigned int lds = (unsigned int)(unsigned long long)lds_dst;
  unsigned long long ga = (unsigned long long)gsrc;
  asm volatile("global_load_async_to_lds_b128 %0, %1, off"
               :: "v"(lds), "v"(ga) : "memory");
}
__device__ __forceinline__ void async_wait0() {
  asm volatile("s_wait_asynccnt 0x0" ::: "memory");
}

// ---------------------------------------------------------------------------
// Kernel A: K path. Pool 64 keys (mean+max) -> [256] vec, matvec with wk[kh],
// RoPE over gh, store f16. One 128-thread block per (nb, kh, b).
// ---------------------------------------------------------------------------
__global__ void ag_kpool_kernel(const float* __restrict__ k,
                                const float* __restrict__ wk,
                                const float* __restrict__ cosk,
                                const float* __restrict__ sink,
                                _Float16* __restrict__ kp) {
  __shared__ float kc[256];
  __shared__ float kpv[128];
  const int nb = blockIdx.x, kh = blockIdx.y, b = blockIdx.z;
  const int tid = threadIdx.x;                     // 0..127 = d / gh index
  const float* base = k + (((size_t)b * SEQ + (size_t)nb * BLKK) * HK + kh) * DH + tid;
  float mean = 0.f, mx = -3.4e38f;
  #pragma unroll 4
  for (int j = 0; j < BLKK; ++j) {
    float v = base[(size_t)j * HK * DH];           // coalesced across tid
    mean += v; mx = fmaxf(mx, v);
  }
  kc[tid]       = mean * (1.0f / BLKK);
  kc[128 + tid] = mx;
  __syncthreads();
  const float* wkh = wk + (size_t)kh * 256 * GH + tid;
  float acc = 0.f;
  #pragma unroll 8
  for (int i = 0; i < 256; ++i) acc += kc[i] * wkh[(size_t)i * GH];
  kpv[tid] = acc;
  __syncthreads();
  const float partner = (tid < 64) ? -kpv[tid + 64] : kpv[tid - 64];
  const size_t cb = ((size_t)b * NB + nb) * GH + tid;
  const float o = acc * cosk[cb] + partner * sink[cb];
  kp[(((size_t)b * HK + kh) * NB + nb) * GH + tid] = (_Float16)o;
}

// ---------------------------------------------------------------------------
// Kernel B: Q projection GEMM via WMMA f16->f32, RoPE epilogue in-register.
// Block = 256 thr (8 waves). Tile: M=128 (16 rows/wave), N=128, K=512.
// LDS: wq[kh] transposed to [n][k] f16 + q tile [m][k] f16 (2*130KB).
// Per K-step: gather A + ALL 8 B frags first, then 8 WMMAs (latency overlap).
// ---------------------------------------------------------------------------
__global__ void ag_qproj_kernel(const float* __restrict__ q,
                                const float* __restrict__ wq,
                                const float* __restrict__ cosq,
                                const float* __restrict__ sinq,
                                _Float16* __restrict__ qp) {
  extern __shared__ _Float16 smem[];
  _Float16* wqT = smem;                    // [128 n][A_STR k]
  _Float16* Aq  = smem + 128 * A_STR;      // [128 m][A_STR k]
  const int mblk = blockIdx.x, kh = blockIdx.y, b = blockIdx.z;
  const int tid = threadIdx.x;
  const int m0 = mblk * 128;

  // Stage wq[kh] (512x128, N-contiguous) -> transposed f16 [n][k]
  const float* wqk = wq + (size_t)kh * 512 * GH;
  for (int it = 0; it < 64; ++it) {
    int idx = it * 256 + tid;              // 16384 float4 total
    int krow = idx >> 5;                   // 32 float4 per K-row
    int c4   = idx & 31;
    const float4 v = ((const float4*)(wqk + (size_t)krow * GH))[c4];
    int n = c4 * 4;
    wqT[(n + 0) * A_STR + krow] = (_Float16)v.x;
    wqT[(n + 1) * A_STR + krow] = (_Float16)v.y;
    wqT[(n + 2) * A_STR + krow] = (_Float16)v.z;
    wqT[(n + 3) * A_STR + krow] = (_Float16)v.w;
  }
  // Stage q rows m0..m0+127: each row = 512 contiguous floats (heads kh*4..+3)
  for (int it = 0; it < 64; ++it) {
    int idx = it * 256 + tid;              // 16384 float4 total
    int r  = idx >> 7;                     // 128 float4 per row
    int c4 = idx & 127;
    const float* src = q + (((size_t)b * SEQ + (m0 + r)) * HQ + kh * GRP) * DH;
    const float4 v = ((const float4*)src)[c4];
    _Float16* dst = Aq + r * A_STR + c4 * 4;
    dst[0] = (_Float16)v.x; dst[1] = (_Float16)v.y;
    dst[2] = (_Float16)v.z; dst[3] = (_Float16)v.w;
  }
  __syncthreads();

  const int lane = tid & 31, wave = tid >> 5;
  const int lr0 = wave * 16;
  const int col = lane & 15, half = lane >> 4;

  v8f acc[8] = {};
  for (int ks = 0; ks < 16; ++ks) {
    const int k0 = ks * 32;
    // A frag: 16x32 f16. lanes0-15 K=[0..7,16..23], lanes16-31 K=[8..15,24..31]
    Frag Af;
    {
      const _Float16* ar = Aq + (lr0 + col) * A_STR + k0 + half * 8;
      #pragma unroll
      for (int r = 0; r < 8; ++r) Af.h[r]     = ar[r];
      #pragma unroll
      for (int r = 0; r < 8; ++r) Af.h[8 + r] = ar[16 + r];
    }
    // Gather all 8 B frags before any WMMA so ds_loads overlap WMMA issue.
    Frag Bf[8];
    #pragma unroll
    for (int t = 0; t < 8; ++t) {
      const _Float16* br = wqT + (t * 16 + col) * A_STR + k0 + half * 8;
      #pragma unroll
      for (int r = 0; r < 16; ++r) Bf[t].h[r] = br[r < 8 ? r : r + 8];
    }
    #pragma unroll
    for (int t = 0; t < 8; ++t) {
      acc[t] = __builtin_amdgcn_wmma_f32_16x16x32_f16(
          false, Af.v, false, Bf[t].v, (short)0, acc[t], false, false);
    }
  }

  // RoPE epilogue: col n and n+64 are same lane, same VGPR, tiles t and t+4.
  const size_t outbase = ((size_t)b * HK + kh) * SEQ;
  #pragma unroll
  for (int r = 0; r < 8; ++r) {
    const int s = m0 + lr0 + half * 8 + r;
    const float* cq = cosq + ((size_t)b * SEQ + s) * GH;
    const float* sq = sinq + ((size_t)b * SEQ + s) * GH;
    _Float16* outp = qp + (outbase + s) * GH;
    #pragma unroll
    for (int t = 0; t < 4; ++t) {
      const int gl = t * 16 + col, gh2 = gl + 64;
      const float xl = acc[t][r], xh = acc[t + 4][r];
      outp[gl]  = (_Float16)(xl * cq[gl]  - xh * sq[gl]);
      outp[gh2] = (_Float16)(xh * cq[gh2] + xl * sq[gh2]);
    }
  }
}

// ---------------------------------------------------------------------------
// Kernel C: scores qp . kp^T via WMMA (M=128/blk, N=128, K=128), scale + mask,
// row softmax via half-wave shuffles (a C-tile row lives in one 16-lane half).
// Tile staging uses CDNA5 async global->LDS B128 copies (ASYNCcnt tracked).
// ---------------------------------------------------------------------------
__global__ void ag_score_kernel(const _Float16* __restrict__ qp,
                                const _Float16* __restrict__ kp,
                                const float* __restrict__ mask,
                                float* __restrict__ out) {
  extern __shared__ _Float16 smem[];
  _Float16* qpt = smem;                    // [128 m][Q_STR k]
  _Float16* kpt = smem + 128 * Q_STR;      // [128 n][Q_STR k]
  const int sblk = blockIdx.x, kh = blockIdx.y, b = blockIdx.z;
  const int tid = threadIdx.x;
  const int s0 = sblk * 128;

  // Stage kp[b,kh] (128x128 f16) and qp rows (128x128 f16) via async B128
  // copies: each row = 16 chunks of 16B; LDS rows are 272B (16B aligned).
  const _Float16* kpsrc = kp + (((size_t)b * HK + kh) * NB) * GH;
  const _Float16* qpsrc = qp + (((size_t)b * HK + kh) * SEQ + s0) * GH;
  #pragma unroll
  for (int it = 0; it < 8; ++it) {
    int idx = it * 256 + tid;              // 2048 chunks per tile
    int row = idx >> 4, c = idx & 15;      // 16 x 16B per row
    async_copy_b128(qpt + row * Q_STR + c * 8, qpsrc + (size_t)row * GH + c * 8);
    async_copy_b128(kpt + row * Q_STR + c * 8, kpsrc + (size_t)row * GH + c * 8);
  }
  async_wait0();
  __syncthreads();

  const int lane = tid & 31, wave = tid >> 5;
  const int lr0 = wave * 16;
  const int col = lane & 15, half = lane >> 4;

  v8f acc[8] = {};
  #pragma unroll
  for (int ks = 0; ks < 4; ++ks) {
    const int k0 = ks * 32;
    Frag Af;
    {
      const _Float16* ar = qpt + (lr0 + col) * Q_STR + k0 + half * 8;
      #pragma unroll
      for (int r = 0; r < 8; ++r) Af.h[r]     = ar[r];
      #pragma unroll
      for (int r = 0; r < 8; ++r) Af.h[8 + r] = ar[16 + r];
    }
    Frag Bf[8];
    #pragma unroll
    for (int t = 0; t < 8; ++t) {
      const _Float16* br = kpt + (t * 16 + col) * Q_STR + k0 + half * 8;
      #pragma unroll
      for (int r = 0; r < 16; ++r) Bf[t].h[r] = br[r < 8 ? r : r + 8];
    }
    #pragma unroll
    for (int t = 0; t < 8; ++t) {
      acc[t] = __builtin_amdgcn_wmma_f32_16x16x32_f16(
          false, Af.v, false, Bf[t].v, (short)0, acc[t], false, false);
    }
  }

  // scale + mask + softmax over n (8 tiles x 16 lanes of this half-wave)
  const float scale = 0.08838834764831845f;  // 1/sqrt(128)
  #pragma unroll
  for (int r = 0; r < 8; ++r) {
    const int s = s0 + lr0 + half * 8 + r;
    const float* mrow = mask + ((size_t)b * SEQ + s) * NB;
    float v[8];
    float mx = -3.4e38f;
    #pragma unroll
    for (int t = 0; t < 8; ++t) {
      v[t] = acc[t][r] * scale + mrow[t * 16 + col];
      mx = fmaxf(mx, v[t]);
    }
    #pragma unroll
    for (int m = 1; m <= 8; m <<= 1) mx = fmaxf(mx, __shfl_xor(mx, m, 32));
    float sum = 0.f;
    #pragma unroll
    for (int t = 0; t < 8; ++t) { v[t] = __expf(v[t] - mx); sum += v[t]; }
    #pragma unroll
    for (int m = 1; m <= 8; m <<= 1) sum += __shfl_xor(sum, m, 32);
    const float inv = 1.0f / sum;
    float* orow = out + (((size_t)b * HK + kh) * SEQ + s) * NB;
    #pragma unroll
    for (int t = 0; t < 8; ++t) orow[t * 16 + col] = v[t] * inv;
  }
}

// ---------------------------------------------------------------------------
extern "C" void kernel_launch(void* const* d_in, const int* in_sizes, int n_in,
                              void* d_out, int out_size, void* d_ws, size_t ws_size,
                              hipStream_t stream) {
  const float* q    = (const float*)d_in[0];
  const float* k    = (const float*)d_in[1];
  const float* mask = (const float*)d_in[2];
  const float* cosq = (const float*)d_in[3];
  const float* sinq = (const float*)d_in[4];
  const float* cosk = (const float*)d_in[5];
  const float* sink = (const float*)d_in[6];
  const float* wq   = (const float*)d_in[7];
  const float* wk   = (const float*)d_in[8];
  float* out = (float*)d_out;

  _Float16* kp = (_Float16*)d_ws;                                  // 512 KB
  _Float16* qp = (_Float16*)((char*)d_ws + (size_t)(1 << 20));     // 32 MB

  ag_kpool_kernel<<<dim3(NB, HK, BATCH), 128, 0, stream>>>(k, wk, cosk, sink, kp);

  const size_t lds_q = (size_t)2 * 128 * A_STR * sizeof(_Float16); // ~260 KB
  ag_qproj_kernel<<<dim3(SEQ / 128, HK, BATCH), 256, lds_q, stream>>>(
      q, wq, cosq, sinq, qp);

  const size_t lds_s = (size_t)2 * 128 * Q_STR * sizeof(_Float16); // ~68 KB
  ag_score_kernel<<<dim3(SEQ / 128, HK, BATCH), 256, lds_s, stream>>>(
      qp, kp, mask, out);
}
